// GCN_37185826848799
// MI455X (gfx1250) — matrix operand order repaired and checked by reading, not compiled, not run on hardware
//
#include <hip/hip_runtime.h>
#include <hip/hip_bf16.h>
#include <math.h>

// ---------------------------------------------------------------------------
// GCN + CRF-attention + LN + log_softmax for MI455X (gfx1250, wave32).
//  - adj (400MB) -> CSR in one pass (everything N x N becomes ~51 nnz/row SpMM)
//  - dense GEMMs via v_wmma_f32_16x16x32_f16, A staged in LDS by the
//    Tensor Data Mover (tensor_load_to_lds + s_wait_tensorcnt, double buffered)
//  - f16 operand copies / transposed weights prepared once so fragment loads
//    are contiguous b128s (no in-loop cvt, no strided global columns)
// ---------------------------------------------------------------------------

#define NNODES 10000
#define NFEAT  512
#define NHID   256
#define NCLASS 16
#define MAXD   128   // nnz/row cap; E[deg]~51, sigma~7 -> overflow prob ~0

typedef __attribute__((ext_vector_type(16))) _Float16 v16h;
typedef __attribute__((ext_vector_type(8)))  _Float16 v8h;
typedef __attribute__((ext_vector_type(8)))  float    v8f;
typedef __attribute__((ext_vector_type(4)))  unsigned int v4u;
typedef __attribute__((ext_vector_type(8)))  int      v8i;
typedef __attribute__((ext_vector_type(4)))  int      v4i;

#if defined(__has_builtin)
#if __has_builtin(__builtin_amdgcn_tensor_load_to_lds)
#define HAVE_TDM 1
#endif
#endif

__device__ __forceinline__ float wred_sum(float v) {
#pragma unroll
  for (int m = 16; m >= 1; m >>= 1) v += __shfl_xor(v, m, 32);
  return v;
}
__device__ __forceinline__ float wred_max(float v) {
#pragma unroll
  for (int m = 16; m >= 1; m >>= 1) v = fmaxf(v, __shfl_xor(v, m, 32));
  return v;
}

#ifdef HAVE_TDM
// 2-D TDM load: tile (tile_d0 x tile_d1) of 2-byte elements from a row-major
// tensor (row stride stride0 elements) at gaddr -> LDS[lds_addr].
// D# packing per CDNA5 ISA.
__device__ __forceinline__ void tdm_load_2d_f16(unsigned lds_addr, const void* gaddr,
                                                unsigned tensor_d0, unsigned tensor_d1,
                                                unsigned long long stride0,
                                                unsigned tile_d0, unsigned tile_d1) {
  unsigned long long ga = (unsigned long long)(uintptr_t)gaddr;
  v4u g0;
  g0[0] = 1u;                                            // count=1, user D#
  g0[1] = lds_addr;                                      // [63:32] lds byte addr
  g0[2] = (unsigned)(ga & 0xFFFFFFFFu);                  // global_addr lo
  g0[3] = (unsigned)((ga >> 32) & 0x01FFFFFFu) | (2u << 30); // ga hi | type=2
  v8i g1;
  g1[0] = (int)(1u << 16);                               // data_size=1 (2B)
  g1[1] = (int)((tensor_d0 & 0xFFFFu) << 16);            // tensor_dim0[15:0]
  g1[2] = (int)(((tensor_d0 >> 16) & 0xFFFFu) | ((tensor_d1 & 0xFFFFu) << 16));
  g1[3] = (int)(((tensor_d1 >> 16) & 0xFFFFu) | ((tile_d0 & 0xFFFFu) << 16));
  g1[4] = (int)(tile_d1 & 0xFFFFu);                      // tile_dim1, tile_dim2=0
  g1[5] = (int)(unsigned)(stride0 & 0xFFFFFFFFull);      // dim0_stride lo
  g1[6] = (int)(unsigned)((stride0 >> 32) & 0xFFFFull);  // dim0_stride hi
  g1[7] = 0;
  v4i z4 = {0, 0, 0, 0};
#if __clang_major__ >= 23
  v8i z8 = {0, 0, 0, 0, 0, 0, 0, 0};
  __builtin_amdgcn_tensor_load_to_lds(g0, g1, z4, z4, z8, 0);
#else
  __builtin_amdgcn_tensor_load_to_lds(g0, g1, z4, z4, 0);
#endif
}
#endif

// ---------------------------------------------------------------------------
// f32 -> f16 elementwise copy (grid-stride)
// ---------------------------------------------------------------------------
__global__ void cvt_f16_kernel(const float* __restrict__ src,
                               _Float16* __restrict__ dst, long n) {
  long i = (long)blockIdx.x * blockDim.x + threadIdx.x;
  const long step = (long)gridDim.x * blockDim.x;
  for (; i < n; i += step) dst[i] = (_Float16)src[i];
}

// f32 [R,C] -> f16 transposed [C,R]
__global__ void transpose_cvt_kernel(const float* __restrict__ src,
                                     _Float16* __restrict__ dst, int R, int C) {
  long i = (long)blockIdx.x * blockDim.x + threadIdx.x;
  const long n = (long)R * C;
  const long step = (long)gridDim.x * blockDim.x;
  for (; i < n; i += step) {
    const int r = (int)(i / C), c = (int)(i % C);
    dst[(size_t)c * R + r] = (_Float16)src[i];
  }
}

// ---------------------------------------------------------------------------
// 1) CSR build: one wave/row, ordered ballot compaction (deterministic,
//    column-sorted). The only pass that touches the 400MB dense adj.
// ---------------------------------------------------------------------------
__global__ void build_csr_kernel(const float* __restrict__ adj,
                                 int* __restrict__ nnz,
                                 int* __restrict__ col,
                                 float* __restrict__ val) {
  const int wid  = (int)((blockIdx.x * blockDim.x + threadIdx.x) >> 5);
  const int lane = (int)(threadIdx.x & 31);
  if (wid >= NNODES) return;
  const float* row = adj + (size_t)wid * NNODES;
  int count = 0;
  for (int c0 = 0; c0 < NNODES; c0 += 32) {
    const int c = c0 + lane;
    const float v = (c < NNODES) ? row[c] : 0.0f;
    const bool p = v > 0.0f;
    const unsigned mask = __builtin_amdgcn_ballot_w32(p);
    const int prefix = __popc(mask & ((1u << lane) - 1u));
    if (p) {
      const int idx = count + prefix;
      if (idx < MAXD) {
        col[(size_t)wid * MAXD + idx] = c;
        val[(size_t)wid * MAXD + idx] = v;
      }
    }
    count += __popc(mask);
  }
  if (lane == 0) nnz[wid] = (count < MAXD) ? count : MAXD;
}

// ---------------------------------------------------------------------------
// 2) WMMA GEMM: C[M,N] = A16[M,K] @ BT16[N,K]^T  (f16 in, f32 accumulate).
//    Block = 8 waves; wave w owns m-tile mblock*8+w, all share n-tile tn.
//    A strip (128 x 32 k-slice) staged in LDS by TDM, double buffered.
//    B fragment = one contiguous 32B load per lane from transposed weights.
// ---------------------------------------------------------------------------
__global__ void gemm_wmma_kernel(const _Float16* __restrict__ A16,
                                 const _Float16* __restrict__ BT16,
                                 float* __restrict__ C,
                                 int M, int N, int K) {
  __shared__ _Float16 abuf[2][128 * 32];  // 16KB
  const int wave = (int)(threadIdx.x >> 5);
  const int lane = (int)(threadIdx.x & 31);
  const int tilesM = M >> 4;
  const int mblocks = (tilesM + 7) >> 3;
  const int mblock = (int)(blockIdx.x % mblocks);
  const int tn     = (int)(blockIdx.x / mblocks);
  const int mtile  = mblock * 8 + wave;
  const bool active = mtile < tilesM;     // wave-uniform

  const int mrow  = lane & 15;
  const int half  = lane >> 4;
  const int kgrpA = half * 16;
  const int kgrpB = half * 16;
  const int m0 = mblock * 128;

  const _Float16* Bp = BT16 + (size_t)(tn * 16 + mrow) * K;  // lane's BT row
  const _Float16* Ag = A16 + (size_t)m0 * K;

  const int nk = K >> 5;
  const unsigned ldsbase = (unsigned)(uintptr_t)(&abuf[0][0]);
  (void)ldsbase;

#ifdef HAVE_TDM
  if (wave == 0) {
    tdm_load_2d_f16(ldsbase, Ag, (unsigned)K, (unsigned)(M - m0),
                    (unsigned long long)K, 32u, 128u);
    __builtin_amdgcn_s_wait_tensorcnt(0);
  }
  __syncthreads();
#else
  {  // cooperative fallback staging of slice 0
    const int t = (int)threadIdx.x;
    const int r = t >> 1, cc = (t & 1) * 16;
    v8h z = {};
    v8h u0 = z, u1 = z;
    if (m0 + r < M) {
      u0 = *(const v8h*)(Ag + (size_t)r * K + cc);
      u1 = *(const v8h*)(Ag + (size_t)r * K + cc + 8);
    }
    *(v8h*)(&abuf[0][r * 32 + cc]) = u0;
    *(v8h*)(&abuf[0][r * 32 + cc + 8]) = u1;
  }
  __syncthreads();
#endif

  v8f acc = {};
  for (int ks = 0; ks < nk; ++ks) {
    const int nxt = ks + 1;
#ifdef HAVE_TDM
    if (wave == 0 && nxt < nk) {
      tdm_load_2d_f16(ldsbase + (unsigned)((nxt & 1) * 128 * 32 * 2),
                      Ag + (size_t)nxt * 32, (unsigned)K, (unsigned)(M - m0),
                      (unsigned long long)K, 32u, 128u);
    }
#else
    if (nxt < nk) {
      const int t = (int)threadIdx.x;
      const int r = t >> 1, cc = (t & 1) * 16;
      v8h z = {};
      v8h u0 = z, u1 = z;
      if (m0 + r < M) {
        u0 = *(const v8h*)(Ag + (size_t)r * K + (size_t)nxt * 32 + cc);
        u1 = *(const v8h*)(Ag + (size_t)r * K + (size_t)nxt * 32 + cc + 8);
      }
      *(v8h*)(&abuf[nxt & 1][r * 32 + cc]) = u0;
      *(v8h*)(&abuf[nxt & 1][r * 32 + cc + 8]) = u1;
    }
#endif
    if (active) {
      const _Float16* as = &abuf[ks & 1][(wave * 16 + mrow) * 32];
      const v8h alo = *(const v8h*)(as + kgrpA);        // K = kgrpA..+7
      const v8h ahi = *(const v8h*)(as + kgrpA + 8);    // K = kgrpA+8..+15
      v16h a;
#pragma unroll
      for (int i = 0; i < 8; ++i) { a[i] = alo[i]; a[i + 8] = ahi[i]; }
      if (nxt < nk) __builtin_prefetch(Bp + (size_t)nxt * 32 + kgrpB, 0, 1);
      const v16h b = *(const v16h*)(Bp + (size_t)ks * 32 + kgrpB);  // contiguous K
      acc = __builtin_amdgcn_wmma_f32_16x16x32_f16(
          false, a, false, b, (short)0, acc, false, false);
    }
#ifdef HAVE_TDM
    if (wave == 0 && nxt < nk) __builtin_amdgcn_s_wait_tensorcnt(0);
#endif
    __syncthreads();
  }
  if (active) {
#pragma unroll
    for (int r = 0; r < 8; ++r) {
      const int row = mtile * 16 + r + half * 8;
      C[(size_t)row * N + tn * 16 + mrow] = acc[r];
    }
  }
}

// ---------------------------------------------------------------------------
// 3) H = relu(SpMM(adj, T1) + b1); also emits f16 copy for the Q/K GEMMs.
// ---------------------------------------------------------------------------
__global__ void spmm_relu_kernel(const int* __restrict__ nnz,
                                 const int* __restrict__ col,
                                 const float* __restrict__ val,
                                 const float* __restrict__ Tin,
                                 const float* __restrict__ b1,
                                 float* __restrict__ H,
                                 _Float16* __restrict__ H16) {
  const int wid  = (int)((blockIdx.x * blockDim.x + threadIdx.x) >> 5);
  const int lane = (int)(threadIdx.x & 31);
  if (wid >= NNODES) return;
  float acc[8] = {0, 0, 0, 0, 0, 0, 0, 0};
  const int n = nnz[wid];
  const int*   cp = col + (size_t)wid * MAXD;
  const float* vp = val + (size_t)wid * MAXD;
  for (int e = 0; e < n; ++e) {
    const int   c = cp[e];
    const float w = vp[e];
    const float4* t = (const float4*)(Tin + (size_t)c * NHID + lane * 8);
    const float4 u0 = t[0], u1 = t[1];
    acc[0] += w * u0.x; acc[1] += w * u0.y; acc[2] += w * u0.z; acc[3] += w * u0.w;
    acc[4] += w * u1.x; acc[5] += w * u1.y; acc[6] += w * u1.z; acc[7] += w * u1.w;
  }
#pragma unroll
  for (int i = 0; i < 8; ++i) {
    const int f = lane * 8 + i;
    const float h = fmaxf(acc[i] + b1[f], 0.0f);
    H[(size_t)wid * NHID + f] = h;
    H16[(size_t)wid * NHID + f] = (_Float16)h;
  }
}

// ---------------------------------------------------------------------------
// 4) Edge-masked softmax attention. Non-edges get -1e9 => exp()==0 exactly in
//    f32, and every row has a self-loop, so edge-only softmax is exact.
//    Scores live in registers: lane (e&31) owns slot e>>5.
// ---------------------------------------------------------------------------
__global__ void edge_attn_kernel(const int* __restrict__ nnz,
                                 const int* __restrict__ col,
                                 const float* __restrict__ Q,
                                 const float* __restrict__ Km,
                                 float* __restrict__ attn) {
  const int wid  = (int)((blockIdx.x * blockDim.x + threadIdx.x) >> 5);
  const int lane = (int)(threadIdx.x & 31);
  if (wid >= NNODES) return;
  const int n = nnz[wid];
  const float4* q4 = (const float4*)(Q + (size_t)wid * NHID + lane * 8);
  const float4 q0 = q4[0], q1 = q4[1];

  float sloc[MAXD / 32];
#pragma unroll
  for (int t = 0; t < MAXD / 32; ++t) sloc[t] = -1e30f;

  for (int e = 0; e < n; ++e) {
    const int c = col[(size_t)wid * MAXD + e];
    const float4* k4 = (const float4*)(Km + (size_t)c * NHID + lane * 8);
    const float4 k0 = k4[0], k1 = k4[1];
    float p = q0.x * k0.x + q0.y * k0.y + q0.z * k0.z + q0.w * k0.w +
              q1.x * k1.x + q1.y * k1.y + q1.z * k1.z + q1.w * k1.w;
    p = wred_sum(p) * 0.0625f;  // 1/sqrt(NHID)
    if ((e & 31) == lane) sloc[e >> 5] = p;
  }
  float mx = sloc[0];
#pragma unroll
  for (int t = 1; t < MAXD / 32; ++t) mx = fmaxf(mx, sloc[t]);
  mx = wred_max(mx);

  float s = 0.0f;
#pragma unroll
  for (int t = 0; t < MAXD / 32; ++t) {
    const int e = t * 32 + lane;
    if (e < n) { sloc[t] = expf(sloc[t] - mx); s += sloc[t]; } else sloc[t] = 0.0f;
  }
  s = wred_sum(s);
  const float inv = 1.0f / s;
#pragma unroll
  for (int t = 0; t < MAXD / 32; ++t) {
    const int e = t * 32 + lane;
    if (e < n) attn[(size_t)wid * MAXD + e] = sloc[t] * inv;
  }
}

// ---------------------------------------------------------------------------
// 5) CRF mean-field iteration: out = (a*H + b*(attn@prev))/(a+b).
//    Last iteration fuses LayerNorm and emits the f16 operand for out@W2.
// ---------------------------------------------------------------------------
__global__ void crf_iter_kernel(const int* __restrict__ nnz,
                                const int* __restrict__ col,
                                const float* __restrict__ attn,
                                const float* __restrict__ H,
                                const float* __restrict__ prev,
                                float* __restrict__ outp,
                                const float* __restrict__ alphap,
                                const float* __restrict__ betap,
                                int do_ln,
                                const float* __restrict__ lnw,
                                const float* __restrict__ lnb,
                                _Float16* __restrict__ out16) {
  const int wid  = (int)((blockIdx.x * blockDim.x + threadIdx.x) >> 5);
  const int lane = (int)(threadIdx.x & 31);
  if (wid >= NNODES) return;
  const float alpha = *alphap, beta = *betap;
  const float invab = 1.0f / (alpha + beta);
  const int n = nnz[wid];

  float acc[8] = {0, 0, 0, 0, 0, 0, 0, 0};
  for (int e = 0; e < n; ++e) {
    const int   c = col[(size_t)wid * MAXD + e];
    const float w = attn[(size_t)wid * MAXD + e];
    const float4* p4 = (const float4*)(prev + (size_t)c * NHID + lane * 8);
    const float4 u0 = p4[0], u1 = p4[1];
    acc[0] += w * u0.x; acc[1] += w * u0.y; acc[2] += w * u0.z; acc[3] += w * u0.w;
    acc[4] += w * u1.x; acc[5] += w * u1.y; acc[6] += w * u1.z; acc[7] += w * u1.w;
  }
  float o[8];
#pragma unroll
  for (int i = 0; i < 8; ++i) {
    const float h = H[(size_t)wid * NHID + lane * 8 + i];
    o[i] = (alpha * h + beta * acc[i]) * invab;
  }
  if (do_ln) {
    float s = 0.0f, s2 = 0.0f;
#pragma unroll
    for (int i = 0; i < 8; ++i) { s += o[i]; s2 += o[i] * o[i]; }
    s  = wred_sum(s);
    s2 = wred_sum(s2);
    const float mu  = s * (1.0f / NHID);
    const float var = s2 * (1.0f / NHID) - mu * mu;
    const float r   = rsqrtf(var + 1e-5f);
#pragma unroll
    for (int i = 0; i < 8; ++i) {
      const int f = lane * 8 + i;
      const float v = (o[i] - mu) * r * lnw[f] + lnb[f];
      out16[(size_t)wid * NHID + f] = (_Float16)v;
    }
  } else {
#pragma unroll
    for (int i = 0; i < 8; ++i) outp[(size_t)wid * NHID + lane * 8 + i] = o[i];
  }
}

// ---------------------------------------------------------------------------
// 6) y = SpMM(adj, Z) + b2, then log_softmax over NCLASS=16.
// ---------------------------------------------------------------------------
__global__ void final_out_kernel(const int* __restrict__ nnz,
                                 const int* __restrict__ col,
                                 const float* __restrict__ val,
                                 const float* __restrict__ Z,
                                 const float* __restrict__ b2,
                                 float* __restrict__ out) {
  const int wid  = (int)((blockIdx.x * blockDim.x + threadIdx.x) >> 5);
  const int lane = (int)(threadIdx.x & 31);
  if (wid >= NNODES) return;
  const int n = nnz[wid];
  const int c16  = lane & 15;
  const int half = lane >> 4;
  float y = 0.0f;
  for (int e = half; e < n; e += 2) {
    const int   c = col[(size_t)wid * MAXD + e];
    const float w = val[(size_t)wid * MAXD + e];
    y += w * Z[(size_t)c * NCLASS + c16];
  }
  y += __shfl_xor(y, 16, 32);  // both halves now hold the full sum
  y += b2[c16];
  float mx = y;
#pragma unroll
  for (int m = 8; m >= 1; m >>= 1) mx = fmaxf(mx, __shfl_xor(mx, m, 32));
  const float ex = expf(y - mx);
  float s = ex;
#pragma unroll
  for (int m = 8; m >= 1; m >>= 1) s += __shfl_xor(s, m, 32);
  const float ls = y - mx - logf(s);
  if (half == 0) out[(size_t)wid * NCLASS + c16] = ls;
}

// ---------------------------------------------------------------------------
// launch
// ---------------------------------------------------------------------------
extern "C" void kernel_launch(void* const* d_in, const int* in_sizes, int n_in,
                              void* d_out, int out_size, void* d_ws, size_t ws_size,
                              hipStream_t stream) {
  const float* x     = (const float*)d_in[0];
  const float* adj   = (const float*)d_in[1];
  const float* W1    = (const float*)d_in[2];
  const float* b1    = (const float*)d_in[3];
  const float* Wq    = (const float*)d_in[4];
  const float* Wk    = (const float*)d_in[5];
  const float* alpha = (const float*)d_in[6];
  const float* beta  = (const float*)d_in[7];
  const float* lnw   = (const float*)d_in[8];
  const float* lnb   = (const float*)d_in[9];
  const float* W2    = (const float*)d_in[10];
  const float* b2    = (const float*)d_in[11];
  float* out = (float*)d_out;

  char* ws = (char*)d_ws;
  const size_t SZ_MAT = (size_t)NNODES * NHID * sizeof(float);      // 10.24 MB
  float*    T1    = (float*)(ws + 0 * SZ_MAT);
  float*    H     = (float*)(ws + 1 * SZ_MAT);
  float*    Qm    = (float*)(ws + 2 * SZ_MAT);
  float*    Km    = (float*)(ws + 3 * SZ_MAT);
  float*    O0    = (float*)(ws + 4 * SZ_MAT);
  float*    O1    = (float*)(ws + 5 * SZ_MAT);
  char*     p     = ws + 6 * SZ_MAT;
  _Float16* X16   = (_Float16*)p;  p += (size_t)NNODES * NFEAT * 2;   // 10.24 MB
  _Float16* H16   = (_Float16*)p;  p += (size_t)NNODES * NHID * 2;    //  5.12 MB
  _Float16* OL16  = (_Float16*)p;  p += (size_t)NNODES * NHID * 2;    //  5.12 MB
  _Float16* W1T   = (_Float16*)p;  p += (size_t)NFEAT * NHID * 2;     //  256 KB
  _Float16* WqT   = (_Float16*)p;  p += (size_t)NHID * NHID * 2;
  _Float16* WkT   = (_Float16*)p;  p += (size_t)NHID * NHID * 2;
  _Float16* W2T   = (_Float16*)p;  p += (size_t)NHID * NCLASS * 2;
  float*    Z     = (float*)p;     p += (size_t)NNODES * NCLASS * 4;
  int*      nnz   = (int*)p;       p += (size_t)NNODES * 4;
  int*      col   = (int*)p;       p += (size_t)NNODES * MAXD * 4;
  float*    val   = (float*)p;     p += (size_t)NNODES * MAXD * 4;
  float*    att   = (float*)p;     // + NNODES*MAXD*4   (total ~98 MB)

  const int BLK = 256;                                  // 8 waves / block
  const int rowBlocks = (NNODES * 32 + BLK - 1) / BLK;  // one wave per row
  const int tilesM  = NNODES / 16;                      // 625
  const int mblocks = (tilesM + 7) / 8;                 // 79

  // operand prep (tiny, bandwidth-trivial)
  cvt_f16_kernel<<<2048, BLK, 0, stream>>>(x, X16, (long)NNODES * NFEAT);
  transpose_cvt_kernel<<<512, BLK, 0, stream>>>(W1, W1T, NFEAT, NHID);
  transpose_cvt_kernel<<<256, BLK, 0, stream>>>(Wq, WqT, NHID, NHID);
  transpose_cvt_kernel<<<256, BLK, 0, stream>>>(Wk, WkT, NHID, NHID);
  transpose_cvt_kernel<<<16,  BLK, 0, stream>>>(W2, W2T, NHID, NCLASS);

  // 1) adjacency -> CSR (the single 400MB pass)
  build_csr_kernel<<<rowBlocks, BLK, 0, stream>>>(adj, nnz, col, val);

  // 2) T1 = x @ W1 (WMMA + TDM-staged A)
  gemm_wmma_kernel<<<mblocks * (NHID / 16), BLK, 0, stream>>>(
      X16, W1T, T1, NNODES, NHID, NFEAT);
  // 3) H = relu(adj @ T1 + b1)
  spmm_relu_kernel<<<rowBlocks, BLK, 0, stream>>>(nnz, col, val, T1, b1, H, H16);
  // 4) Q = H @ Wq ; K = H @ Wk
  gemm_wmma_kernel<<<mblocks * (NHID / 16), BLK, 0, stream>>>(
      H16, WqT, Qm, NNODES, NHID, NHID);
  gemm_wmma_kernel<<<mblocks * (NHID / 16), BLK, 0, stream>>>(
      H16, WkT, Km, NNODES, NHID, NHID);
  // 5) edge-masked softmax attention
  edge_attn_kernel<<<rowBlocks, BLK, 0, stream>>>(nnz, col, Qm, Km, att);
  // 6) 3 CRF iterations; LN fused into the last (emits f16 operand)
  crf_iter_kernel<<<rowBlocks, BLK, 0, stream>>>(nnz, col, att, H, H,  O0,
                                                 alpha, beta, 0, lnw, lnb, OL16);
  crf_iter_kernel<<<rowBlocks, BLK, 0, stream>>>(nnz, col, att, H, O0, O1,
                                                 alpha, beta, 0, lnw, lnb, OL16);
  crf_iter_kernel<<<rowBlocks, BLK, 0, stream>>>(nnz, col, att, H, O1, O0,
                                                 alpha, beta, 1, lnw, lnb, OL16);
  // 7) Z = LN(out) @ W2 (single 16-wide n-tile)
  gemm_wmma_kernel<<<mblocks * (NCLASS / 16), BLK, 0, stream>>>(
      OL16, W2T, Z, NNODES, NCLASS, NHID);
  // 8) out = log_softmax(adj @ Z + b2)
  final_out_kernel<<<rowBlocks, BLK, 0, stream>>>(nnz, col, val, Z, b2, out);
}
